// DrugGNN_89197880803531
// MI455X (gfx1250) — compile-verified, matrix operand order checked by reading, not run on hardware
//
#include <hip/hip_runtime.h>
#include <hip/hip_bf16.h>
#include <cstdint>

// ---------------------------------------------------------------------------
// Types for CDNA5 WMMA (wave32): v_wmma_f32_16x16x32_bf16
// ---------------------------------------------------------------------------
typedef __attribute__((ext_vector_type(16))) __bf16 v16bf;
typedef __attribute__((ext_vector_type(8)))  float  v8f;

union FragU {              // 32 bytes: two 16B LDS loads -> one WMMA operand
    uint4 q[2];
    v16bf v;
};

// ---------------------------------------------------------------------------
// gfx1250 async copy: global memory -> LDS, 16B per lane, ASYNCcnt-tracked
// ---------------------------------------------------------------------------
__device__ __forceinline__ void async_copy_b128(unsigned lds_off, const void* gptr) {
    asm volatile("global_load_async_to_lds_b128 %0, %1, off"
                 :: "v"(lds_off), "v"((unsigned long long)(uintptr_t)gptr)
                 : "memory");
}
__device__ __forceinline__ void wait_async0() {
    asm volatile("s_wait_asynccnt 0x0" ::: "memory");
}
__device__ __forceinline__ unsigned lds_addr32(const void* p) {
    return (unsigned)(uintptr_t)p;   // flat shared aperture: offset in addr[31:0]
}

// ---------------------------------------------------------------------------
// Small helpers
// ---------------------------------------------------------------------------
__device__ __forceinline__ unsigned short f2bf(float f) {
    unsigned u = __float_as_uint(f);
    unsigned r = u + 0x7FFFu + ((u >> 16) & 1u);   // round-to-nearest-even
    return (unsigned short)(r >> 16);
}

// order-preserving float<->uint encoding for atomicMax on floats
__device__ __forceinline__ unsigned fenc(float f) {
    unsigned u = __float_as_uint(f);
    return (u & 0x80000000u) ? ~u : (u | 0x80000000u);
}
__device__ __forceinline__ float fdec(unsigned e) {
    unsigned u = (e & 0x80000000u) ? (e & 0x7FFFFFFFu) : ~e;
    return __uint_as_float(u);
}

__global__ void fill_u32_kernel(unsigned* __restrict__ p, unsigned v, long n) {
    long i = (long)blockIdx.x * blockDim.x + threadIdx.x;
    if (i < n) p[i] = v;
}

// ---------------------------------------------------------------------------
// fp32 -> bf16 converters (activations padded; weights transposed+padded)
// ---------------------------------------------------------------------------
__global__ void cvt_act_bf16_kernel(const float* __restrict__ src, int cols,
                                    unsigned short* __restrict__ dst, int kpad,
                                    long n_elems) {
    long i = (long)blockIdx.x * blockDim.x + threadIdx.x;
    if (i >= n_elems) return;
    long r = i / kpad;
    int  k = (int)(i - r * kpad);
    float v = (k < cols) ? src[r * cols + k] : 0.0f;
    dst[i] = f2bf(v);
}

__global__ void cvt_wT_bf16_kernel(const float* __restrict__ W, int din, int dout,
                                   unsigned short* __restrict__ Wt, int kpad,
                                   long n_elems) {
    long i = (long)blockIdx.x * blockDim.x + threadIdx.x;
    if (i >= n_elems) return;
    long n = i / kpad;
    int  k = (int)(i - n * kpad);
    float v = (k < din) ? W[(long)k * dout + n] : 0.0f;
    Wt[i] = f2bf(v);
}

// ---------------------------------------------------------------------------
// Tiled WMMA GEMM:  C[M,Nout] = A[M,K](bf16) * Wt[Nout,K](bf16)^T, f32 accum
// Block = 256 threads (8 waves). Tile 128(M) x 64(N), K-step 32.
// Double-buffered LDS staged via GLOBAL_LOAD_ASYNC_TO_LDS_B128 (ASYNCcnt):
// issue next K-tile's copies, compute WMMA on current tile, one barrier/iter.
// Out-of-range M rows are *clamped* (their C rows are never stored), so the
// staging path is branch-free.
// ---------------------------------------------------------------------------
#define TM 128
#define TN 64
#define TK 32
#define LSTR 40   // padded LDS row stride in shorts (80B) -> no bank conflicts

__global__ __launch_bounds__(256)
void gemm_bf16_wmma_kernel(const unsigned short* __restrict__ A, int lda,
                           const unsigned short* __restrict__ Bt, int ldb,
                           float* __restrict__ C, int ldc,
                           int M, int K) {
    __shared__ unsigned short lA[2][TM * LSTR];   // 2 x 10240 B
    __shared__ unsigned short lB[2][TN * LSTR];   // 2 x  5120 B

    const int tid  = threadIdx.x;
    const int wave = tid >> 5;        // 0..7 -> 16-row strip of the M tile
    const int lane = tid & 31;
    const int mBase = blockIdx.x * TM;
    const int nBase = blockIdx.y * TN;

    v8f acc0 = {}, acc1 = {}, acc2 = {}, acc3 = {};

    const int r0 = tid >> 2;          // 0..63
    const int cg = (tid & 3) * 8;     // 0,8,16,24 (ushort col group; 16B)

    const int mrow = wave * 16 + (lane & 15);
    const int kb   = (lane >> 4) * 8; // K split per 16-bit A-matrix layout
    const int nrow = lane & 15;

    // issue async stage of K-tile k0 into LDS buffer bufi (3x16B per thread)
    auto stage = [&](int bufi, int k0) {
#pragma unroll
        for (int rr = 0; rr < 2; ++rr) {          // A: rows r0, r0+64
            int row = r0 + rr * 64;
            int gm  = mBase + row;
            if (gm >= M) gm = M - 1;              // clamp: row never stored
            async_copy_b128(lds_addr32(&lA[bufi][row * LSTR + cg]),
                            &A[(long)gm * lda + k0 + cg]);
        }
        async_copy_b128(lds_addr32(&lB[bufi][r0 * LSTR + cg]),   // B: row r0
                        &Bt[(long)(nBase + r0) * ldb + k0 + cg]);
    };

    int buf = 0;
    stage(0, 0);

    for (int k0 = 0; k0 < K; k0 += TK) {
        wait_async0();          // my async copies into `buf` are complete
        __syncthreads();        // everyone's copies visible; prev reads done

        if (k0 + TK < K) stage(buf ^ 1, k0 + TK);   // prefetch next tile

        // -------- load fragments from LDS buffer `buf` --------------------
        const unsigned short* la = lA[buf];
        const unsigned short* lb = lB[buf];
        FragU af;
        af.q[0] = *(const uint4*)&la[mrow * LSTR + kb];
        af.q[1] = *(const uint4*)&la[mrow * LSTR + kb + 16];

        FragU bf0, bf1, bf2, bf3;
        bf0.q[0] = *(const uint4*)&lb[(nrow +  0) * LSTR + kb];
        bf0.q[1] = *(const uint4*)&lb[(nrow +  0) * LSTR + kb + 16];
        bf1.q[0] = *(const uint4*)&lb[(nrow + 16) * LSTR + kb];
        bf1.q[1] = *(const uint4*)&lb[(nrow + 16) * LSTR + kb + 16];
        bf2.q[0] = *(const uint4*)&lb[(nrow + 32) * LSTR + kb];
        bf2.q[1] = *(const uint4*)&lb[(nrow + 32) * LSTR + kb + 16];
        bf3.q[0] = *(const uint4*)&lb[(nrow + 48) * LSTR + kb];
        bf3.q[1] = *(const uint4*)&lb[(nrow + 48) * LSTR + kb + 16];

        acc0 = __builtin_amdgcn_wmma_f32_16x16x32_bf16(false, af.v, false, bf0.v,
                                                       (short)0, acc0, false, false);
        acc1 = __builtin_amdgcn_wmma_f32_16x16x32_bf16(false, af.v, false, bf1.v,
                                                       (short)0, acc1, false, false);
        acc2 = __builtin_amdgcn_wmma_f32_16x16x32_bf16(false, af.v, false, bf2.v,
                                                       (short)0, acc2, false, false);
        acc3 = __builtin_amdgcn_wmma_f32_16x16x32_bf16(false, af.v, false, bf3.v,
                                                       (short)0, acc3, false, false);
        buf ^= 1;
    }

    // -------- store C tile (16x16 f32 layout: VGPR r -> M = r + 8*(lane>>4))
    const int crow = wave * 16 + 8 * (lane >> 4);
    const int ccol = nBase + (lane & 15);
#pragma unroll
    for (int r = 0; r < 8; ++r) {
        int gm = mBase + crow + r;
        if (gm < M) {
            float* c = &C[(long)gm * ldc + ccol];
            c[0]  = acc0[r];
            c[16] = acc1[r];
            c[32] = acc2[r];
            c[48] = acc3[r];
        }
    }
}

// ---------------------------------------------------------------------------
// Per-(node,head) attention logits: one wave32 per (node,head), C=128
// ---------------------------------------------------------------------------
template <int HEADS>
__global__ void logits_kernel(const float* __restrict__ h,
                              const float* __restrict__ a_src,
                              const float* __restrict__ a_dst,
                              float* __restrict__ als, float* __restrict__ ald,
                              int n_nodes) {
    long gid  = (long)blockIdx.x * blockDim.x + threadIdx.x;
    long wid  = gid >> 5;
    int  lane = (int)(gid & 31);
    if (wid >= (long)n_nodes * HEADS) return;
    int node = (int)(wid / HEADS);
    int hd   = (int)(wid - (long)node * HEADS);

    const float4 hv = *(const float4*)&h[(long)node * HEADS * 128 + hd * 128 + lane * 4];
    const float4 sa = *(const float4*)&a_src[hd * 128 + lane * 4];
    const float4 da = *(const float4*)&a_dst[hd * 128 + lane * 4];
    float s = hv.x * sa.x + hv.y * sa.y + hv.z * sa.z + hv.w * sa.w;
    float d = hv.x * da.x + hv.y * da.y + hv.z * da.z + hv.w * da.w;
#pragma unroll
    for (int m = 16; m >= 1; m >>= 1) {
        s += __shfl_xor(s, m, 32);
        d += __shfl_xor(d, m, 32);
    }
    if (lane == 0) { als[wid] = s; ald[wid] = d; }
}

// ---------------------------------------------------------------------------
// Edge pass 1: e = leaky_relu(als[src]+ald[dst]); store; atomicMax per (dst,h)
// Self-loops appended implicitly: edge id e in [E, E+N) -> src=dst=e-E.
// ---------------------------------------------------------------------------
template <int HEADS>
__global__ void edge_max_kernel(const int* __restrict__ ei, int E, int etot,
                                const float* __restrict__ als,
                                const float* __restrict__ ald,
                                float* __restrict__ pbuf,
                                unsigned* __restrict__ nmax) {
    long idx = (long)blockIdx.x * blockDim.x + threadIdx.x;
    if (idx >= (long)etot * HEADS) return;
    long e  = idx / HEADS;
    int  hd = (int)(idx - e * HEADS);
    int s, d;
    if (e < E) { s = ei[e]; d = ei[E + e]; } else { s = d = (int)(e - E); }
    float v = als[(long)s * HEADS + hd] + ald[(long)d * HEADS + hd];
    v = (v > 0.0f) ? v : 0.2f * v;                 // leaky relu, slope 0.2
    pbuf[idx] = v;
    atomicMax(&nmax[(long)d * HEADS + hd], fenc(v));
}

// Edge pass 2: p = exp(e - max[dst]); store; atomicAdd denominator per (dst,h)
template <int HEADS>
__global__ void edge_exp_kernel(const int* __restrict__ ei, int E, int etot,
                                const unsigned* __restrict__ nmax,
                                float* __restrict__ pbuf,
                                float* __restrict__ nden) {
    long idx = (long)blockIdx.x * blockDim.x + threadIdx.x;
    if (idx >= (long)etot * HEADS) return;
    long e  = idx / HEADS;
    int  hd = (int)(idx - e * HEADS);
    int d = (e < E) ? ei[E + e] : (int)(e - E);
    float mx = fdec(nmax[(long)d * HEADS + hd]);
    float p  = __expf(pbuf[idx] - mx);
    pbuf[idx] = p;
    atomicAdd(&nden[(long)d * HEADS + hd], p);
}

// Edge pass 3: out[dst] += alpha * h[src], float4-vectorized atomic scatter
template <int HEADS>
__global__ void edge_agg_kernel(const int* __restrict__ ei, int E, int etot,
                                const float* __restrict__ h,
                                const float* __restrict__ pbuf,
                                const float* __restrict__ nden,
                                float* __restrict__ out) {
    constexpr int HC  = HEADS * 128;
    constexpr int CPE = HC >> 2;                   // float4 chunks per edge
    long idx = (long)blockIdx.x * blockDim.x + threadIdx.x;
    if (idx >= (long)etot * CPE) return;
    long e  = idx / CPE;                           // compile-time shift
    int  r  = (int)(idx - e * CPE);
    int  c4 = r * 4;
    int  hd = c4 >> 7;                             // C = 128
    int s, d;
    if (e < E) { s = ei[e]; d = ei[E + e]; } else { s = d = (int)(e - E); }
    float alpha = pbuf[e * HEADS + hd] /
                  (nden[(long)d * HEADS + hd] + 1e-16f);
    float4 hv = *(const float4*)&h[(long)s * HC + c4];
    float* o  = &out[(long)d * HC + c4];
    atomicAdd(&o[0], hv.x * alpha);
    atomicAdd(&o[1], hv.y * alpha);
    atomicAdd(&o[2], hv.z * alpha);
    atomicAdd(&o[3], hv.w * alpha);
}

// ---------------------------------------------------------------------------
// Fused bias + BatchNorm(eval) + optional ReLU
// ---------------------------------------------------------------------------
__global__ void bias_bn_kernel(const float* __restrict__ in, int dout,
                               const float* __restrict__ b,
                               const float* __restrict__ g,
                               const float* __restrict__ bt,
                               const float* __restrict__ rm,
                               const float* __restrict__ rv,
                               float* __restrict__ out, int do_relu, long n) {
    long i = (long)blockIdx.x * blockDim.x + threadIdx.x;
    if (i >= n) return;
    int col = (int)(i % dout);
    float v = in[i] + b[col];
    float y = (v - rm[col]) * g[col] * rsqrtf(rv[col] + 1e-5f) + bt[col];
    if (do_relu) y = fmaxf(y, 0.0f);
    out[i] = y;
}

// ---------------------------------------------------------------------------
// Graph pooling: segment sum/count/max over batch
// ---------------------------------------------------------------------------
__global__ void pool_scatter_kernel(const float* __restrict__ h,
                                    const int* __restrict__ batch,
                                    float* __restrict__ gsum,
                                    unsigned* __restrict__ gmax,
                                    float* __restrict__ gcnt, long n_nodes) {
    long idx = (long)blockIdx.x * blockDim.x + threadIdx.x;
    if (idx >= n_nodes * 128) return;
    long n = idx >> 7;
    int  c = (int)(idx & 127);
    int  g = batch[n];
    float v = h[idx];
    atomicAdd(&gsum[(long)g * 128 + c], v);
    atomicMax(&gmax[(long)g * 128 + c], fenc(v));
    if (c == 0) atomicAdd(&gcnt[g], 1.0f);
}

__global__ void pool_final_kernel(const float* __restrict__ gsum,
                                  const unsigned* __restrict__ gmax,
                                  const float* __restrict__ gcnt,
                                  float* __restrict__ out, int G) {
    long idx = (long)blockIdx.x * blockDim.x + threadIdx.x;
    if (idx >= (long)G * 128) return;
    int g = (int)(idx >> 7);
    int c = (int)(idx & 127);
    float cnt = gcnt[g];
    float mean = (cnt > 0.0f) ? gsum[idx] / cnt : 0.0f;
    float mx   = (cnt > 0.0f) ? fdec(gmax[idx]) : 0.0f;   // guard empty graphs
    out[(long)g * 256 + c]       = mean;
    out[(long)g * 256 + 128 + c] = mx;
}

// ---------------------------------------------------------------------------
// Host launch
// ---------------------------------------------------------------------------
static inline dim3 blk1(long n, int t = 256) { return dim3((unsigned)((n + t - 1) / t)); }

extern "C" void kernel_launch(void* const* d_in, const int* in_sizes, int n_in,
                              void* d_out, int out_size, void* d_ws, size_t ws_size,
                              hipStream_t stream) {
    const float* x     = (const float*)d_in[0];
    const int*   ei    = (const int*)d_in[1];
    const int*   batch = (const int*)d_in[2];

    const int N    = in_sizes[2];
    const int E    = in_sizes[1] / 2;
    const int FIN  = in_sizes[0] / N;
    const int Etot = E + N;
    const int G    = out_size / 256;

    const float* W[3]  = {(const float*)d_in[3],  (const float*)d_in[11], (const float*)d_in[19]};
    const float* As[3] = {(const float*)d_in[4],  (const float*)d_in[12], (const float*)d_in[20]};
    const float* Ad[3] = {(const float*)d_in[5],  (const float*)d_in[13], (const float*)d_in[21]};
    const float* Bb[3] = {(const float*)d_in[6],  (const float*)d_in[14], (const float*)d_in[22]};
    const float* Gg[3] = {(const float*)d_in[7],  (const float*)d_in[15], (const float*)d_in[23]};
    const float* Bt[3] = {(const float*)d_in[8],  (const float*)d_in[16], (const float*)d_in[24]};
    const float* Rm[3] = {(const float*)d_in[9],  (const float*)d_in[17], (const float*)d_in[25]};
    const float* Rv[3] = {(const float*)d_in[10], (const float*)d_in[18], (const float*)d_in[26]};

    const int Din [3] = {FIN, 512, 512};
    const int Dout[3] = {512, 512, 128};
    const int Hh  [3] = {4, 4, 1};
    const int Kp  [3] = {(FIN + 31) & ~31, 512, 512};

    // ---- workspace carve-out ------------------------------------------------
    char*  ws  = (char*)d_ws;
    size_t off = 0;
    auto carve = [&](size_t bytes) {
        void* p = ws + off;
        off = (off + bytes + 255) & ~(size_t)255;
        return p;
    };
    unsigned short* Abf = (unsigned short*)carve((size_t)N * 512 * 2);   // bf16 activations
    unsigned short* Wbf = (unsigned short*)carve((size_t)512 * 512 * 2); // bf16 Wt
    float*    Hbuf = (float*)carve((size_t)N * 512 * 4);  // GEMM out / layer out
    float*    Agg  = (float*)carve((size_t)N * 512 * 4);  // attention aggregate
    float*    als  = (float*)carve((size_t)N * 4 * 4);
    float*    ald  = (float*)carve((size_t)N * 4 * 4);
    unsigned* nmax = (unsigned*)carve((size_t)N * 4 * 4);
    float*    nden = (float*)carve((size_t)N * 4 * 4);
    float*    pbuf = (float*)carve((size_t)Etot * 4 * 4);
    float*    gsum = (float*)carve((size_t)G * 128 * 4);
    unsigned* gmax = (unsigned*)carve((size_t)G * 128 * 4);
    float*    gcnt = (float*)carve((size_t)G * 4);

    const float* layer_in = x;   // fp32 input of current layer

    for (int l = 0; l < 3; ++l) {
        const int din = Din[l], dout = Dout[l], heads = Hh[l], kp = Kp[l];

        // 1) convert activations + transposed weights to bf16 (zero-padded K)
        long na = (long)N * kp;
        cvt_act_bf16_kernel<<<blk1(na), 256, 0, stream>>>(layer_in, din, Abf, kp, na);
        long nw = (long)dout * kp;
        cvt_wT_bf16_kernel<<<blk1(nw), 256, 0, stream>>>(W[l], din, dout, Wbf, kp, nw);

        // 2) WMMA GEMM: Hbuf[N,dout] = A * W
        dim3 ggrid((N + TM - 1) / TM, dout / TN);
        gemm_bf16_wmma_kernel<<<ggrid, 256, 0, stream>>>(Abf, kp, Wbf, kp,
                                                         Hbuf, dout, N, kp);

        // 3) per-(node,head) logits
        long nwaves = (long)N * heads;
        if (heads == 4)
            logits_kernel<4><<<blk1(nwaves * 32), 256, 0, stream>>>(Hbuf, As[l], Ad[l],
                                                                    als, ald, N);
        else
            logits_kernel<1><<<blk1(nwaves * 32), 256, 0, stream>>>(Hbuf, As[l], Ad[l],
                                                                    als, ald, N);

        // 4) zero segment-softmax state + aggregate buffer
        long nh = (long)N * heads;
        fill_u32_kernel<<<blk1(nh), 256, 0, stream>>>(nmax, 0u, nh);
        fill_u32_kernel<<<blk1(nh), 256, 0, stream>>>((unsigned*)nden, 0u, nh);
        long nag = (long)N * dout;
        fill_u32_kernel<<<blk1(nag), 256, 0, stream>>>((unsigned*)Agg, 0u, nag);

        // 5) edge softmax + scatter
        long neh  = (long)Etot * heads;
        long nagg = (long)Etot * (heads * 128 / 4);
        if (heads == 4) {
            edge_max_kernel<4><<<blk1(neh), 256, 0, stream>>>(ei, E, Etot, als, ald,
                                                              pbuf, nmax);
            edge_exp_kernel<4><<<blk1(neh), 256, 0, stream>>>(ei, E, Etot, nmax,
                                                              pbuf, nden);
            edge_agg_kernel<4><<<blk1(nagg), 256, 0, stream>>>(ei, E, Etot, Hbuf,
                                                               pbuf, nden, Agg);
        } else {
            edge_max_kernel<1><<<blk1(neh), 256, 0, stream>>>(ei, E, Etot, als, ald,
                                                              pbuf, nmax);
            edge_exp_kernel<1><<<blk1(neh), 256, 0, stream>>>(ei, E, Etot, nmax,
                                                              pbuf, nden);
            edge_agg_kernel<1><<<blk1(nagg), 256, 0, stream>>>(ei, E, Etot, Hbuf,
                                                               pbuf, nden, Agg);
        }

        // 6) bias + BN (+ReLU for layers 0,1); write back into Hbuf
        bias_bn_kernel<<<blk1(nag), 256, 0, stream>>>(Agg, dout, Bb[l], Gg[l],
                                                      Bt[l], Rm[l], Rv[l],
                                                      Hbuf, (l < 2) ? 1 : 0, nag);
        layer_in = Hbuf;
    }

    // ---- pooling: mean + max over graphs -> [G, 256] ------------------------
    long ngc = (long)G * 128;
    fill_u32_kernel<<<blk1(ngc), 256, 0, stream>>>((unsigned*)gsum, 0u, ngc);
    fill_u32_kernel<<<blk1(ngc), 256, 0, stream>>>(gmax, 0u, ngc);
    fill_u32_kernel<<<blk1(G), 256, 0, stream>>>((unsigned*)gcnt, 0u, G);

    long npool = (long)N * 128;
    pool_scatter_kernel<<<blk1(npool), 256, 0, stream>>>(Hbuf, batch, gsum, gmax,
                                                         gcnt, N);
    pool_final_kernel<<<blk1(ngc), 256, 0, stream>>>(gsum, gmax, gcnt,
                                                     (float*)d_out, G);
}
